// Encoder_89386859364908
// MI455X (gfx1250) — matrix operand order repaired and checked by reading, not compiled, not run on hardware
//
#include <hip/hip_runtime.h>
#include <math.h>

// Problem constants
#define BB   64
#define TT   256
#define EE   256
#define HH   512
#define G4H  2048           // 4*H
#define BT   (BB * TT)
#define NWG_SCAN 32         // cooperative workgroups in the scan kernel

typedef __attribute__((ext_vector_type(16))) __bf16 v16bf;
typedef __attribute__((ext_vector_type(8)))  float  v8f;

typedef union {
    v16bf        v;
    unsigned int u[8];
    uint4        q[2];
} Frag;

__device__ __forceinline__ unsigned short f2bf(float f) {
    unsigned int u = __float_as_uint(f);
    u += 0x7FFFu + ((u >> 16) & 1u);          // round-to-nearest-even
    return (unsigned short)(u >> 16);
}
__device__ __forceinline__ float bf2f(unsigned short h) {
    return __uint_as_float(((unsigned int)h) << 16);
}
__device__ __forceinline__ float sigmoidf_(float x) {
    return 1.0f / (1.0f + __expf(-x));
}

// D = A(16x32 bf16) x B(32x16 bf16) + C(16x16 f32), wave32 WMMA
__device__ __forceinline__ v8f wmma_bf16(const Frag& a, const Frag& b, v8f c) {
    return __builtin_amdgcn_wmma_f32_16x16x32_bf16(
        false, a.v, false, b.v, (short)0, c, false, false);
}

// Load a 16x32 bf16 A-fragment from a hoisted per-lane row pointer.
// arow = base + mrow*stride + 8*(lane>=16); per ISA 7.12.2 the lane's 16
// elements live at dword offsets {0..3, 8..11} from (arow + kt*32 elems).
__device__ __forceinline__ void load_a_kt(Frag& a, const unsigned short* arow,
                                          int kt) {
    const unsigned int* p = (const unsigned int*)(arow + (kt << 5));
    a.u[0] = p[0];  a.u[1] = p[1];  a.u[2] = p[2];  a.u[3] = p[3];
    a.u[4] = p[8];  a.u[5] = p[9];  a.u[6] = p[10]; a.u[7] = p[11];
}

// Generation-counter grid barrier (all NWG workgroups co-resident).
__device__ __forceinline__ void grid_barrier(unsigned int* cnt, int nwg,
                                             unsigned int& gen) {
    __syncthreads();
    if (threadIdx.x == 0) {
        __threadfence();
        atomicAdd(cnt, 1u);
        const unsigned int tgt = (gen + 1u) * (unsigned int)nwg;
        while (__hip_atomic_load(cnt, __ATOMIC_ACQUIRE,
                                 __HIP_MEMORY_SCOPE_AGENT) < tgt) {
            __builtin_amdgcn_s_sleep(1);
        }
    }
    gen += 1u;
    __syncthreads();
}

// ---------------------------------------------------------------------------
// init: zero the grid-barrier counters (runs every call -> graph-replay safe)
// ---------------------------------------------------------------------------
__global__ void init_counters(unsigned int* c) {
    if (threadIdx.x < 16) c[threadIdx.x] = 0;
}

// ---------------------------------------------------------------------------
// Embedding gather: e16[bt, k] = bf16(emb[x[bt], k]),  E = 256
// ---------------------------------------------------------------------------
__global__ __launch_bounds__(256) void embed_kernel(
    const int* __restrict__ x, const float* __restrict__ emb,
    unsigned short* __restrict__ e16) {
    const size_t i  = (size_t)blockIdx.x * 256 + threadIdx.x;   // < BT*EE exact
    const int    bt = (int)(i >> 8);
    const int    k  = (int)(i & 255);
    e16[i] = f2bf(emb[(size_t)x[bt] * EE + k]);
}

// ---------------------------------------------------------------------------
// Pack a row-major f32 [K,N] weight into WMMA B-operand tiles (bf16).
// Tile p = nt*(K/32)+kt holds 32 lanes x 16 bf16 contiguous.
// Lane l: n = nt*16 + (l&15); K pair base = kt*32 + 8*(l>=16) + 2*(v&3) + 16*(v>>2).
// ---------------------------------------------------------------------------
__global__ __launch_bounds__(256) void pack_b_kernel(
    const float* __restrict__ src, unsigned short* __restrict__ dst,
    int K, int N) {
    const int id    = blockIdx.x * 256 + threadIdx.x;  // one dword (2 bf16)
    const int total = (K * N) >> 1;
    if (id >= total) return;
    const int p   = id >> 8;
    const int rem = id & 255;
    const int l   = rem >> 3;
    const int v   = rem & 7;
    const int KT  = K >> 5;
    const int nt  = p / KT;
    const int kt  = p - nt * KT;
    const int k   = (kt << 5) + ((l >> 4) << 3) + ((v & 3) << 1) + ((v >> 2) << 4);
    const int n   = (nt << 4) + (l & 15);
    const unsigned int lo = f2bf(src[(size_t)k * N + n]);
    const unsigned int hi = f2bf(src[(size_t)(k + 1) * N + n]);
    ((unsigned int*)dst)[id] = lo | (hi << 16);
}

// ---------------------------------------------------------------------------
// C_bf16[M,N] = A_bf16[M,K] @ Bpacked + bias_f32[N]
// WG = 256 threads (8 waves) computes a 64x128 block; wave = 1 m-tile x 4 n-tiles.
// Software-pipelined: next-kt fragments prefetched before current wmmas.
// ---------------------------------------------------------------------------
__global__ __launch_bounds__(256) void gemm_bias_bf16(
    const unsigned short* __restrict__ A,
    const unsigned short* __restrict__ Bpk,
    const float* __restrict__ bias,
    unsigned short* __restrict__ Cout,
    int M, int K, int N) {
    const int tid     = threadIdx.x;
    const int wave    = tid >> 5;
    const int lane    = tid & 31;
    const int nBlocks = N >> 7;
    const int mblk    = blockIdx.x / nBlocks;
    const int nblk    = blockIdx.x % nBlocks;
    const int mt      = (mblk << 2) + (wave & 3);
    const int ntbase  = (nblk << 3) + ((wave >> 2) << 2);
    const int KT      = K >> 5;
    const int mrow    = (mt << 4) + (lane & 15);

    // Hoisted per-lane bases: A row pointer and per-nt B tile pointers.
    const unsigned short* arow = A + (size_t)mrow * K + ((lane >> 4) << 3);
    const uint4* bbase[4];
#pragma unroll
    for (int j = 0; j < 4; ++j)
        bbase[j] = ((const uint4*)Bpk) +
                   ((size_t)(ntbase + j) * KT << 6) + (lane << 1);

    v8f acc[4];
#pragma unroll
    for (int j = 0; j < 4; ++j)
        acc[j] = (v8f){0.f, 0.f, 0.f, 0.f, 0.f, 0.f, 0.f, 0.f};

    Frag a_cur, b_cur[4];
    load_a_kt(a_cur, arow, 0);
#pragma unroll
    for (int j = 0; j < 4; ++j) {
        b_cur[j].q[0] = bbase[j][0];
        b_cur[j].q[1] = bbase[j][1];
    }

    for (int kt = 0; kt < KT; ++kt) {
        const int ktn = (kt + 1 < KT) ? (kt + 1) : kt;  // uniform: reload last
        Frag a_nxt, b_nxt[4];
        load_a_kt(a_nxt, arow, ktn);
#pragma unroll
        for (int j = 0; j < 4; ++j) {
            const uint4* bp = bbase[j] + (ktn << 6);
            b_nxt[j].q[0] = bp[0];
            b_nxt[j].q[1] = bp[1];
        }
#pragma unroll
        for (int j = 0; j < 4; ++j)
            acc[j] = wmma_bf16(a_cur, b_cur[j], acc[j]);
        a_cur = a_nxt;
#pragma unroll
        for (int j = 0; j < 4; ++j) b_cur[j] = b_nxt[j];
    }

    const int mro = (mt << 4) + ((lane >> 4) << 3);   // C layout: M = r + 8*(lane>=16)
#pragma unroll
    for (int j = 0; j < 4; ++j) {
        const int   n  = ((ntbase + j) << 4) + (lane & 15);
        const float bn = bias[n];
#pragma unroll
        for (int r = 0; r < 8; ++r) {
            Cout[(size_t)(mro + r) * N + n] = f2bf(acc[j][r] + bn);
        }
    }
}

// ---------------------------------------------------------------------------
// Persistent cooperative LSTM scan (one direction).
//  - 32 WGs, WG owns 16 hidden units -> 4 gate n-tiles; U slice in LDS (64KB).
//  - per step: z = h_prev @ U (WMMA) + xs[t]; gates; c in LDS; h ping-pong in
//    global bf16; grid barrier per step.  K-loop software-pipelined.
//  - forward (state_out != null): writes raw h to yF, state=h+c at last step.
//  - backward: reads yF, writes 0.5*(yF + h) to ymerge16 (bf16) or ymerge32.
// ---------------------------------------------------------------------------
__global__ __launch_bounds__(256) void lstm_scan(
    const unsigned short* __restrict__ xs,      // [BT, 4H] bf16
    const unsigned short* __restrict__ Upk,     // packed [128 nt][16 kt] tiles
    const float* __restrict__ h0,               // [B,H]
    const float* __restrict__ c0,               // [B,H]
    unsigned short* __restrict__ hping,         // [B,H] bf16
    unsigned short* __restrict__ hpong,         // [B,H] bf16
    float* __restrict__ yF,                     // [B,T,H] f32
    unsigned short* __restrict__ ymerge16,      // bwd L1 out (or null)
    float* __restrict__ ymerge32,               // bwd L2 out (or null)
    float* __restrict__ state_out,              // fwd out h+c (or null)
    unsigned int* __restrict__ barcnt,
    int reverse) {
    extern __shared__ char smem[];
    unsigned int* ub = (unsigned int*)smem;                    // 64KB U slice
    float*        zb = (float*)(smem + 65536);                 // 16KB z buffer
    float*        cb = (float*)(smem + 65536 + 16384);         // 4KB c state

    const int tid  = threadIdx.x;
    const int wave = tid >> 5;
    const int lane = tid & 31;
    const int wg   = blockIdx.x;
    const int jb   = wg << 4;            // hidden base for this WG

    // Stage this WG's 4 gate n-tiles of U (global ntile = g*32 + wg).
    const unsigned int* updw = (const unsigned int*)Upk;
    for (int i = tid; i < 16384; i += 256) {
        const int tl = i >> 8;           // local tile = g*16 + kt
        const int g  = tl >> 4;
        const int kt = tl & 15;
        ub[i] = updw[(((((g << 5) + wg) << 4) + kt) << 8) + (i & 255)];
    }
    // Init c (LDS) and this WG's slice of h (ping buffer, bf16).
    for (int i = tid; i < 1024; i += 256) {
        const int b = i >> 4, jj = i & 15;
        cb[i] = c0[(size_t)b * HH + jb + jj];
        hping[(size_t)b * HH + jb + jj] = f2bf(h0[(size_t)b * HH + jb + jj]);
    }

    unsigned int gen = 0;
    grid_barrier(barcnt, NWG_SCAN, gen);

    const int mt   = wave & 3;
    const int g0   = (wave >> 2) << 1;         // gate tiles {g0, g0+1}
    const int mrow = (mt << 4) + (lane & 15);
    const int mro  = (mt << 4) + ((lane >> 4) << 3);
    const int njj  = lane & 15;

    // Hoisted LDS tile bases for the two gate n-tiles this wave owns.
    const uint4* b0base = ((const uint4*)ub) + (g0 << 10) + (lane << 1);
    const uint4* b1base = ((const uint4*)ub) + ((g0 + 1) << 10) + (lane << 1);

    for (int s = 0; s < TT; ++s) {
        const int t = reverse ? (TT - 1 - s) : s;
        const unsigned short* hprev = (s & 1) ? hpong : hping;
        unsigned short*       hnext = (s & 1) ? hping : hpong;
        const unsigned short* arow  = hprev + (size_t)mrow * HH +
                                      ((lane >> 4) << 3);

        v8f acc0 = (v8f){0.f, 0.f, 0.f, 0.f, 0.f, 0.f, 0.f, 0.f};
        v8f acc1 = (v8f){0.f, 0.f, 0.f, 0.f, 0.f, 0.f, 0.f, 0.f};

        Frag a_cur, b0_cur, b1_cur;
        load_a_kt(a_cur, arow, 0);
        b0_cur.q[0] = b0base[0]; b0_cur.q[1] = b0base[1];
        b1_cur.q[0] = b1base[0]; b1_cur.q[1] = b1base[1];

#pragma unroll 4
        for (int kt = 0; kt < 16; ++kt) {
            const int ktn = (kt < 15) ? (kt + 1) : kt;   // uniform: reload last
            Frag a_nxt, b0_nxt, b1_nxt;
            load_a_kt(a_nxt, arow, ktn);
            {
                const uint4* p0 = b0base + (ktn << 6);
                const uint4* p1 = b1base + (ktn << 6);
                b0_nxt.q[0] = p0[0]; b0_nxt.q[1] = p0[1];
                b1_nxt.q[0] = p1[0]; b1_nxt.q[1] = p1[1];
            }
            acc0 = wmma_bf16(a_cur, b0_cur, acc0);
            acc1 = wmma_bf16(a_cur, b1_cur, acc1);
            a_cur = a_nxt;
            b0_cur = b0_nxt;
            b1_cur = b1_nxt;
        }

        // z = h@U + xs  ->  LDS z-buffer [gate][b][jj]
#pragma unroll
        for (int r = 0; r < 8; ++r) {
            const int    brow = mro + r;
            const size_t xoff = ((size_t)brow * TT + t) * G4H + jb + njj;
            const float  z0   = acc0[r] + bf2f(xs[xoff + (size_t)g0 * HH]);
            const float  z1   = acc1[r] + bf2f(xs[xoff + (size_t)(g0 + 1) * HH]);
            zb[(((g0 << 6) + brow) << 4) + njj]       = z0;
            zb[((((g0 + 1) << 6) + brow) << 4) + njj] = z1;
        }
        __syncthreads();

        // Gates + state update: 1024 (b, jj) elements, 4 per thread.
        const int tnext = reverse ? (t - 1) : (t + 1);
        for (int i = 0; i < 4; ++i) {
            const int   idx = tid + (i << 8);
            const int   b   = idx >> 4, jj = idx & 15;
            const float zi  = zb[(b << 4) + jj];
            const float zf  = zb[((64 + b) << 4) + jj];
            const float zg  = zb[((128 + b) << 4) + jj];
            const float zo  = zb[((192 + b) << 4) + jj];
            const float ig  = sigmoidf_(zi);
            const float fg  = sigmoidf_(zf);
            const float gg  = tanhf(zg);
            const float og  = sigmoidf_(zo);
            const float c   = fg * cb[idx] + ig * gg;
            cb[idx] = c;
            const float h = og * tanhf(c);
            const int   j = jb + jj;
            hnext[(size_t)b * HH + j] = f2bf(h);
            const size_t yoff = ((size_t)b * TT + t) * HH + j;
            if (state_out) {                       // forward direction
                yF[yoff] = h;
                if (s == TT - 1) state_out[(size_t)b * HH + j] = h + c;
            } else {                               // backward: fused average
                const float m = 0.5f * (yF[yoff] + h);
                if (ymerge16) ymerge16[yoff] = f2bf(m);
                else          ymerge32[yoff] = m;
            }
            if (tnext >= 0 && tnext < TT)          // global_prefetch_b8 of next xs
                __builtin_prefetch(&xs[((size_t)b * TT + tnext) * G4H + j], 0, 1);
        }
        grid_barrier(barcnt, NWG_SCAN, gen);
        __threadfence();                           // reader-side visibility of h
    }
}

// ---------------------------------------------------------------------------
extern "C" void kernel_launch(void* const* d_in, const int* in_sizes, int n_in,
                              void* d_out, int out_size, void* d_ws,
                              size_t ws_size, hipStream_t stream) {
    (void)in_sizes; (void)n_in; (void)out_size; (void)ws_size;

    const int*   x   = (const int*)d_in[0];
    const float* h1  = (const float*)d_in[1];   // [4,B,H]
    const float* h2  = (const float*)d_in[2];   // [4,B,H]
    const float* emb = (const float*)d_in[3];
    const float* W1f = (const float*)d_in[4];
    const float* U1f = (const float*)d_in[5];
    const float* b1f = (const float*)d_in[6];
    const float* W1b = (const float*)d_in[7];
    const float* U1b = (const float*)d_in[8];
    const float* b1b = (const float*)d_in[9];
    const float* Wd  = (const float*)d_in[10];
    const float* bd  = (const float*)d_in[11];
    const float* W2f = (const float*)d_in[12];
    const float* U2f = (const float*)d_in[13];
    const float* b2f = (const float*)d_in[14];
    const float* W2b = (const float*)d_in[15];
    const float* U2b = (const float*)d_in[16];
    const float* b2b = (const float*)d_in[17];

    float* outp = (float*)d_out;
    float* y2   = outp;                                   // [B,T,H]
    float* st1  = outp + (size_t)BT * HH;                 // [B,H]
    float* st2  = st1 + (size_t)BB * HH;                  // [B,H]

    char*  wsb = (char*)d_ws;
    size_t off = 0;
    auto bump = [&](size_t bytes) -> char* {
        char* p = wsb + off;
        off += (bytes + 255) & ~(size_t)255;
        return p;
    };
    unsigned int*   bar   = (unsigned int*)bump(256);
    unsigned short* hping = (unsigned short*)bump((size_t)BB * HH * 2);
    unsigned short* hpong = (unsigned short*)bump((size_t)BB * HH * 2);
    unsigned short* e16   = (unsigned short*)bump((size_t)BT * EE * 2);
    unsigned short* U1fp  = (unsigned short*)bump((size_t)HH * G4H * 2);
    unsigned short* U1bp  = (unsigned short*)bump((size_t)HH * G4H * 2);
    unsigned short* U2fp  = (unsigned short*)bump((size_t)HH * G4H * 2);
    unsigned short* U2bp  = (unsigned short*)bump((size_t)HH * G4H * 2);
    unsigned short* W1fp  = (unsigned short*)bump((size_t)EE * G4H * 2);
    unsigned short* W1bp  = (unsigned short*)bump((size_t)EE * G4H * 2);
    unsigned short* W2fp  = (unsigned short*)bump((size_t)HH * G4H * 2);
    unsigned short* W2bp  = (unsigned short*)bump((size_t)HH * G4H * 2);
    unsigned short* Wdp   = (unsigned short*)bump((size_t)HH * HH * 2);
    unsigned short* xsF   = (unsigned short*)bump((size_t)BT * G4H * 2);
    unsigned short* xsB   = (unsigned short*)bump((size_t)BT * G4H * 2);
    float*          yF    = (float*)bump((size_t)BT * HH * 4);
    unsigned short* ym16  = (unsigned short*)bump((size_t)BT * HH * 2);
    unsigned short* z16   = (unsigned short*)bump((size_t)BT * HH * 2);

    init_counters<<<1, 64, 0, stream>>>(bar);
    embed_kernel<<<(BT * EE) / 256, 256, 0, stream>>>(x, emb, e16);

    pack_b_kernel<<<(HH * G4H / 2) / 256, 256, 0, stream>>>(U1f, U1fp, HH, G4H);
    pack_b_kernel<<<(HH * G4H / 2) / 256, 256, 0, stream>>>(U1b, U1bp, HH, G4H);
    pack_b_kernel<<<(HH * G4H / 2) / 256, 256, 0, stream>>>(U2f, U2fp, HH, G4H);
    pack_b_kernel<<<(HH * G4H / 2) / 256, 256, 0, stream>>>(U2b, U2bp, HH, G4H);
    pack_b_kernel<<<(EE * G4H / 2) / 256, 256, 0, stream>>>(W1f, W1fp, EE, G4H);
    pack_b_kernel<<<(EE * G4H / 2) / 256, 256, 0, stream>>>(W1b, W1bp, EE, G4H);
    pack_b_kernel<<<(HH * G4H / 2) / 256, 256, 0, stream>>>(W2f, W2fp, HH, G4H);
    pack_b_kernel<<<(HH * G4H / 2) / 256, 256, 0, stream>>>(W2b, W2bp, HH, G4H);
    pack_b_kernel<<<(HH * HH / 2) / 256, 256, 0, stream>>>(Wd, Wdp, HH, HH);

    const int gXS = (BT / 64) * (G4H / 128);   // 4096 blocks
    const int gDN = (BT / 64) * (HH / 128);    // 1024 blocks
    const size_t lds = 65536 + 16384 + 4096;   // U slice + z + c

    // Layer 1: xs precompute, then fwd/bwd scans
    gemm_bias_bf16<<<gXS, 256, 0, stream>>>(e16, W1fp, b1f, xsF, BT, EE, G4H);
    gemm_bias_bf16<<<gXS, 256, 0, stream>>>(e16, W1bp, b1b, xsB, BT, EE, G4H);
    lstm_scan<<<NWG_SCAN, 256, lds, stream>>>(
        xsF, U1fp, h1, h1 + (size_t)BB * HH, hping, hpong,
        yF, nullptr, nullptr, st1, bar + 0, 0);
    lstm_scan<<<NWG_SCAN, 256, lds, stream>>>(
        xsB, U1bp, h1 + 2 * (size_t)BB * HH, h1 + 3 * (size_t)BB * HH,
        hping, hpong, yF, ym16, nullptr, nullptr, bar + 1, 1);

    // Inter-layer Dense: z = y1 @ Wd + bd
    gemm_bias_bf16<<<gDN, 256, 0, stream>>>(ym16, Wdp, bd, z16, BT, HH, HH);

    // Layer 2: xs precompute (reuse buffers), then fwd/bwd scans
    gemm_bias_bf16<<<gXS, 256, 0, stream>>>(z16, W2fp, b2f, xsF, BT, HH, G4H);
    gemm_bias_bf16<<<gXS, 256, 0, stream>>>(z16, W2bp, b2b, xsB, BT, HH, G4H);
    lstm_scan<<<NWG_SCAN, 256, lds, stream>>>(
        xsF, U2fp, h2, h2 + (size_t)BB * HH, hping, hpong,
        yF, nullptr, nullptr, st2, bar + 2, 0);
    lstm_scan<<<NWG_SCAN, 256, lds, stream>>>(
        xsB, U2bp, h2 + 2 * (size_t)BB * HH, h2 + 3 * (size_t)BB * HH,
        hping, hpong, yF, nullptr, y2, nullptr, bar + 3, 1);
}
